// Model_7696581394855
// MI455X (gfx1250) — matrix operand order repaired and checked by reading, not compiled, not run on hardware
//
#include <hip/hip_runtime.h>
#include <hip/hip_bf16.h>
#include <math.h>

// Problem constants (from reference): B=4, N=512, D=128, O=128, all f32 I/O.
#define B_ 4
#define N_ 512
#define D_ 128
#define O_ 128
#define XSTR 130   // padded row stride in halves (260 B -> dword aligned, conflict-friendly)

typedef _Float16 half16 __attribute__((ext_vector_type(16)));
typedef float    float8 __attribute__((ext_vector_type(8)));

union ABFrag { unsigned int u[8]; half16 h; };

// Branch-free tanh: native gfx1250 v_tanh_f32 (confirmed present), else
// a divergence-free exp-based fallback.
__device__ __forceinline__ float fast_tanh(float v) {
#if __has_builtin(__builtin_amdgcn_tanhf)
  return __builtin_amdgcn_tanhf(v);
#else
  const float ax = __builtin_fabsf(v);
  const float e  = __expf(2.0f * ax);
  const float t  = 1.0f - 2.0f * __builtin_amdgcn_rcpf(e + 1.0f);
  return __builtin_copysignf(t, v);
#endif
}

// Load the 4 K-step B fragments (32 dwords) of one 16-wide o-tile.
// B 32x16 layout: lane column n=lr; lanes<16 hold K {0..7}u{16..23}, lanes>=16 hold K {8..15}u{24..31}.
__device__ __forceinline__ void load_btile(ABFrag bf[4], const _Float16* wst,
                                           int ot, int lr, int half) {
  const unsigned int* brow = (const unsigned int*)(wst + (ot * 16 + lr) * XSTR);
  #pragma unroll
  for (int ks = 0; ks < 4; ++ks) {
    const int k0 = ks * 32 + half * 8;
    #pragma unroll
    for (int p = 0; p < 4; ++p) {
      bf[ks].u[p]     = brow[(k0 >> 1) + p];
      bf[ks].u[4 + p] = brow[((k0 + 16) >> 1) + p];
    }
  }
}

// One workgroup per (b, i). 256 threads = 8 wave32.
// Phase A: stage X_b (f16) and Wscaled^T (f16) into LDS.
// Phase B: S = X_b @ (diag(x_i) Wp) via v_wmma_f32_16x16x32_f16 with
//          double-buffered B fragments, fused tanh(+bp)*aw epilogue,
//          cross-lane reduce -> s[j] scores.
// Phase C: softmax over j, agg = att^T X_b, h = agg@W1 + x_i@W2 + biases,
//          BatchNorm(eval) + SELU, write out[b,i,:].
__global__ __launch_bounds__(256) void fused_pairwise_attn(
    const float* __restrict__ x,   const float* __restrict__ Wp,
    const float* __restrict__ bp,  const float* __restrict__ aw,
    const float* __restrict__ W1,  const float* __restrict__ b1,
    const float* __restrict__ W2,  const float* __restrict__ b2,
    const float* __restrict__ gamma, const float* __restrict__ beta,
    const float* __restrict__ rmean, const float* __restrict__ rvar,
    float* __restrict__ out)
{
  extern __shared__ char smem[];
  _Float16* xs   = (_Float16*)smem;            // [N_][XSTR]  f16 copy of X_b
  _Float16* wst  = xs + N_ * XSTR;             // [O_][XSTR]  (diag(x_i)Wp)^T, o-major
  float*    s_l  = (float*)(wst + O_ * XSTR);  // [N_] attention scores / probs
  float*    red  = s_l + N_;                   // [256] reduction scratch
  float*    xi_l = red + 256;                  // [D_] x[b,i,:]
  float*    bp_l = xi_l + D_;                  // [O_]
  float*    aw_l = bp_l + O_;                  // [O_]
  float*    aggp = aw_l + O_;                  // [256] partial agg, then agg[128]

  const int tid = threadIdx.x;
  const int b   = blockIdx.x >> 9;        // N_ == 512
  const int i   = blockIdx.x & (N_ - 1);
  const float* xb = x + (size_t)b * N_ * D_;

  // ---- Phase A0: small vectors + X_b -> LDS (f16) ----
  if (tid < D_) {
    xi_l[tid] = xb[(size_t)i * D_ + tid];
    bp_l[tid] = bp[tid];
    aw_l[tid] = aw[tid];          // aw is (O,1) flat
  }
  for (int e = tid; e < N_ * D_; e += 256) {
    const int j = e >> 7, d = e & 127;
    xs[j * XSTR + d] = (_Float16)xb[(size_t)j * D_ + d];
  }
  __syncthreads();

  // ---- Phase A1: Wscaled^T[o][d] = x_i[d] * Wp[d][o] (f16) ----
  for (int e = tid; e < D_ * O_; e += 256) {
    const int d = e >> 7, o = e & 127;
    wst[o * XSTR + d] = (_Float16)(xi_l[d] * Wp[d * O_ + o]);
  }
  __syncthreads();

  // ---- Phase B: WMMA GEMM + tanh/aw epilogue ----
  const int wave = tid >> 5;
  const int lane = tid & 31;
  const int half = lane >> 4;   // 0: lanes 0-15, 1: lanes 16-31
  const int lr   = lane & 15;

  for (int jt = 0; jt < 4; ++jt) {
    const int jbase = wave * 64 + jt * 16;

    // A fragments for all 4 K-steps, held across the o-tile loop.
    // f16 A 16x32 layout: lane<16 -> row=lr, K in {0..7}u{16..23}; lane>=16 -> {8..15}u{24..31}
    ABFrag a[4];
    {
      const unsigned int* arow =
          (const unsigned int*)(xs + (jbase + lr) * XSTR);
      #pragma unroll
      for (int ks = 0; ks < 4; ++ks) {
        const int k0 = ks * 32 + half * 8;
        #pragma unroll
        for (int p = 0; p < 4; ++p) {
          a[ks].u[p]     = arow[(k0 >> 1) + p];
          a[ks].u[4 + p] = arow[((k0 + 16) >> 1) + p];
        }
      }
    }

    float acc[8];
    #pragma unroll
    for (int v = 0; v < 8; ++v) acc[v] = 0.f;

    // Double-buffered B fragments: issue next o-tile's 16 ds loads before
    // consuming the current tile, so WMMA never needs a full DS drain.
    ABFrag bf[2][4];
    load_btile(bf[0], wst, 0, lr, half);

    #pragma unroll
    for (int ot = 0; ot < 8; ++ot) {
      const int cur = ot & 1;
      if (ot < 7) load_btile(bf[cur ^ 1], wst, ot + 1, lr, half);

      float8 c = {};
      #pragma unroll
      for (int ks = 0; ks < 4; ++ks)
        c = __builtin_amdgcn_wmma_f32_16x16x32_f16(
                false, a[ks].h, false, bf[cur][ks].h, (short)0, c, false, false);

      // C layout: lane n=lr is column o; VGPR v is row v + 8*half
      const int   o   = ot * 16 + lr;
      const float bpv = bp_l[o];
      const float awv = aw_l[o];
      #pragma unroll
      for (int v = 0; v < 8; ++v) acc[v] += fast_tanh(c[v] + bpv) * awv;
    }

    // Reduce 16 columns per half-wave; lane 0 / lane 16 own rows jbase+v / jbase+8+v
    #pragma unroll
    for (int v = 0; v < 8; ++v) {
      float r = acc[v];
      r += __shfl_xor(r, 1, 32);
      r += __shfl_xor(r, 2, 32);
      r += __shfl_xor(r, 4, 32);
      r += __shfl_xor(r, 8, 32);
      if (lr == 0) s_l[jbase + half * 8 + v] = r;
    }
  }
  __syncthreads();

  // ---- Phase C0: softmax over j (temp = 1.0) ----
  {
    red[tid] = fmaxf(s_l[tid], s_l[tid + 256]);
    __syncthreads();
    for (int off = 128; off > 0; off >>= 1) {
      if (tid < off) red[tid] = fmaxf(red[tid], red[tid + off]);
      __syncthreads();
    }
    const float mx = red[0];
    __syncthreads();
    const float e0 = __expf(s_l[tid] - mx);
    const float e1 = __expf(s_l[tid + 256] - mx);
    s_l[tid] = e0; s_l[tid + 256] = e1;
    red[tid] = e0 + e1;
    __syncthreads();
    for (int off = 128; off > 0; off >>= 1) {
      if (tid < off) red[tid] += red[tid + off];
      __syncthreads();
    }
    const float inv = 1.0f / red[0];
    s_l[tid] *= inv; s_l[tid + 256] *= inv;
  }
  __syncthreads();

  // ---- Phase C1: agg[d] = sum_j att[j] * x[b,j,d] ----
  {
    const int d = tid & 127, part = tid >> 7;
    float ag = 0.f;
    const int j0 = part * 256;
    for (int j = j0; j < j0 + 256; ++j)
      ag += s_l[j] * (float)xs[j * XSTR + d];
    aggp[part * 128 + d] = ag;
  }
  __syncthreads();
  if (tid < 128) aggp[tid] += aggp[128 + tid];
  __syncthreads();

  // ---- Phase C2: projections + BatchNorm(eval) + SELU ----
  if (tid < O_) {
    const int o = tid;
    float h = b1[o] + b2[o];
    for (int d = 0; d < D_; ++d)
      h += aggp[d] * W1[d * O_ + o] + xi_l[d] * W2[d * O_ + o];
    h = (h - rmean[o]) * rsqrtf(rvar[o] + 1e-5f) * gamma[o] + beta[o];
    const float kAlpha = 1.6732632423543772f;
    const float kScale = 1.0507009873554805f;
    h = (h > 0.f) ? kScale * h : kScale * kAlpha * (__expf(h) - 1.f);
    out[((size_t)b * N_ + i) * O_ + o] = h;
  }
}

extern "C" void kernel_launch(void* const* d_in, const int* in_sizes, int n_in,
                              void* d_out, int out_size, void* d_ws, size_t ws_size,
                              hipStream_t stream) {
  (void)in_sizes; (void)n_in; (void)d_ws; (void)ws_size; (void)out_size;

  const float* x     = (const float*)d_in[0];
  const float* Wp    = (const float*)d_in[1];
  const float* bp    = (const float*)d_in[2];
  const float* aw    = (const float*)d_in[3];
  const float* W1    = (const float*)d_in[4];
  const float* b1    = (const float*)d_in[5];
  const float* W2    = (const float*)d_in[6];
  const float* b2    = (const float*)d_in[7];
  const float* gamma = (const float*)d_in[8];
  const float* beta  = (const float*)d_in[9];
  const float* rmean = (const float*)d_in[10];
  const float* rvar  = (const float*)d_in[11];
  float* out = (float*)d_out;

  const size_t shmem =
      (size_t)(N_ * XSTR + O_ * XSTR) * sizeof(_Float16) +
      (size_t)(N_ + 256 + D_ + O_ + O_ + 256) * sizeof(float);

  // Allow >64KB dynamic LDS (CDNA5 WGP has 320KB).
  hipFuncSetAttribute((const void*)fused_pairwise_attn,
                      hipFuncAttributeMaxDynamicSharedMemorySize, (int)shmem);

  dim3 grid(B_ * N_), block(256);
  fused_pairwise_attn<<<grid, block, shmem, stream>>>(
      x, Wp, bp, aw, W1, b1, W2, b2, gamma, beta, rmean, rvar, out);
}
